// GolemModel_39470749450402
// MI455X (gfx1250) — matrix operand order repaired and checked by reading, not compiled, not run on hardware
//
#include <hip/hip_runtime.h>
#include <hip/hip_bf16.h>

// ---- problem constants (match reference) ----
#define E_DIM   1024
#define BATCH   1024
#define H_DIM   64
#define ROWS    128      // batch rows per workgroup (8 waves x 16 rows)
#define LDH     72       // padded LDS row stride in halves (kills bank conflicts)

typedef __attribute__((ext_vector_type(16))) _Float16 v16h;
typedef __attribute__((ext_vector_type(8)))  _Float16 h8;
typedef __attribute__((ext_vector_type(4)))  _Float16 h4;
typedef __attribute__((ext_vector_type(8)))  float    v8f;

__device__ __forceinline__ float sigm(float x) {
    return 1.0f / (1.0f + __expf(-x));
}

__global__ __launch_bounds__(256, 2)
void moe_bmat_kernel(const float* __restrict__ noise,
                     const float* __restrict__ W2,
                     const float* __restrict__ b2,
                     const float* __restrict__ W3,
                     const float* __restrict__ b3,
                     float* __restrict__ out)
{
    __shared__ __align__(16) _Float16 sh_h[ROWS * LDH];   // h = sigmoid(4*noise-2), f16
    __shared__ __align__(16) _Float16 sh_w2t[H_DIM * LDH]; // W2^T (n-major), f16
    __shared__ float sh_b2[H_DIM];
    __shared__ float sh_w3[H_DIM];

    const int tid      = threadIdx.x;
    const int e        = blockIdx.y;
    const int row_base = blockIdx.x * ROWS;

    // ---- stage noise tile -> h (f16, sigmoid applied), fully coalesced b128 loads ----
    const float4* ntile =
        (const float4*)(noise + (size_t)e * (BATCH * H_DIM) + (size_t)row_base * H_DIM);
#pragma unroll
    for (int j = 0; j < 8; ++j) {
        int fi = tid + j * 256;               // 2048 float4 = 128x64 floats
        float4 v = ntile[fi];
        int lin = fi * 4;
        int r = lin >> 6;
        int c = lin & 63;
        h4 hv;
        hv[0] = (_Float16)sigm(4.0f * v.x - 2.0f);
        hv[1] = (_Float16)sigm(4.0f * v.y - 2.0f);
        hv[2] = (_Float16)sigm(4.0f * v.z - 2.0f);
        hv[3] = (_Float16)sigm(4.0f * v.w - 2.0f);
        *(h4*)&sh_h[r * LDH + c] = hv;        // 8B aligned (c % 4 == 0)
    }

    // ---- stage W2^T (f16) ----
    const float4* wt = (const float4*)(W2 + (size_t)e * (H_DIM * H_DIM));
#pragma unroll
    for (int j = 0; j < 4; ++j) {
        int fi = tid + j * 256;               // 1024 float4 = 64x64 floats
        float4 v = wt[fi];
        int lin = fi * 4;
        int k  = lin >> 6;                    // input-feature (K) index
        int n0 = lin & 63;                    // output-feature (N) index
        sh_w2t[(n0 + 0) * LDH + k] = (_Float16)v.x;
        sh_w2t[(n0 + 1) * LDH + k] = (_Float16)v.y;
        sh_w2t[(n0 + 2) * LDH + k] = (_Float16)v.z;
        sh_w2t[(n0 + 3) * LDH + k] = (_Float16)v.w;
    }

    if (tid < 64)       sh_b2[tid]      = b2[(size_t)e * H_DIM + tid];
    else if (tid < 128) sh_w3[tid - 64] = W3[(size_t)e * H_DIM + (tid - 64)];

    __syncthreads();

    // ---- per-wave WMMA: 16 rows x 64 cols, K = 64 ----
    const int wave = tid >> 5;
    const int lane = tid & 31;
    const int m    = lane & 15;   // A row / B col within tile
    const int g    = lane >> 4;   // lane group (K-half select)
    const int arow = wave * 16 + m;

    v8f acc[4] = {};              // 4 N-tiles of 16x16 f32

#pragma unroll
    for (int kk = 0; kk < 2; ++kk) {
        // A fragment: VGPR0-3 = K(8g .. 8g+7), VGPR4-7 = K(16+8g .. 16+8g+7)  [+32*kk]
        h8 alo = *(const h8*)&sh_h[arow * LDH + 8 * g + 32 * kk];
        h8 ahi = *(const h8*)&sh_h[arow * LDH + 16 + 8 * g + 32 * kk];
        v16h af = __builtin_shufflevector(alo, ahi,
                    0, 1, 2, 3, 4, 5, 6, 7, 8, 9, 10, 11, 12, 13, 14, 15);
#pragma unroll
        for (int t = 0; t < 4; ++t) {
            int n = m + 16 * t;
            // B fragment: VGPR0-3 = K(8g .. 8g+7), VGPR4-7 = K(16+8g .. 16+8g+7)  [+32*kk]
            h8 blo = *(const h8*)&sh_w2t[n * LDH + 8 * g + 32 * kk];
            h8 bhi = *(const h8*)&sh_w2t[n * LDH + 16 + 8 * g + 32 * kk];
            v16h bf = __builtin_shufflevector(blo, bhi,
                        0, 1, 2, 3, 4, 5, 6, 7, 8, 9, 10, 11, 12, 13, 14, 15);
            acc[t] = __builtin_amdgcn_wmma_f32_16x16x32_f16(
                         false, af, false, bf, (short)0, acc[t], false, false);
        }
    }

    // ---- epilogue: out[m] = sum_n sigmoid(acc + b2[n]) * W3[n] + b3 ----
    float part[8];
#pragma unroll
    for (int v = 0; v < 8; ++v) part[v] = 0.0f;

#pragma unroll
    for (int t = 0; t < 4; ++t) {
        int n = m + 16 * t;
        float b2v = sh_b2[n];
        float w3v = sh_w3[n];
#pragma unroll
        for (int v = 0; v < 8; ++v)
            part[v] += sigm(acc[t][v] + b2v) * w3v;
    }

    // reduce across the 16 lanes that share the same row group (wave32 masks 1..8)
#pragma unroll
    for (int mask = 1; mask < 16; mask <<= 1) {
#pragma unroll
        for (int v = 0; v < 8; ++v)
            part[v] += __shfl_xor(part[v], mask, 32);
    }

    const float b3v = b3[e];
    if (m == 0) {
        int rbase = row_base + wave * 16 + 8 * g;   // rows 8g..8g+7 of this strip
#pragma unroll
        for (int v = 0; v < 8; ++v)
            out[(size_t)(rbase + v) * E_DIM + e] = part[v] + b3v;   // B_mat[b][e]
    }
}

__global__ void tembed_kernel(const float* __restrict__ T,
                              const float* __restrict__ alpha,
                              const float* __restrict__ beta,
                              const float* __restrict__ bias,
                              float* __restrict__ out)
{
    int i = blockIdx.x * blockDim.x + threadIdx.x;
    if (i < BATCH) {
        out[(size_t)BATCH * E_DIM + i] =
            (1.0f / alpha[0]) * cosf(beta[0] * T[i] + bias[0]);
    }
}

extern "C" void kernel_launch(void* const* d_in, const int* in_sizes, int n_in,
                              void* d_out, int out_size, void* d_ws, size_t ws_size,
                              hipStream_t stream) {
    (void)in_sizes; (void)n_in; (void)out_size; (void)d_ws; (void)ws_size;
    const float* T     = (const float*)d_in[0];
    const float* noise = (const float*)d_in[1];
    // d_in[2] = W1: unused — fc1 output cancels exactly in the reference
    const float* W2    = (const float*)d_in[3];
    const float* b2    = (const float*)d_in[4];
    const float* W3    = (const float*)d_in[5];
    const float* b3    = (const float*)d_in[6];
    const float* alpha = (const float*)d_in[7];
    const float* beta  = (const float*)d_in[8];
    const float* bias  = (const float*)d_in[9];
    float* out = (float*)d_out;

    dim3 grid(BATCH / ROWS, E_DIM);   // (8, 1024): batch tile x expert
    moe_bmat_kernel<<<grid, 256, 0, stream>>>(noise, W2, b2, W3, b3, out);
    tembed_kernel<<<dim3(4), dim3(256), 0, stream>>>(T, alpha, beta, bias, out);
}
